// biAttention_74801150427545
// MI455X (gfx1250) — matrix operand order repaired and checked by reading, not compiled, not run on hardware
//
#include <hip/hip_runtime.h>
#include <math.h>

// ---------------- problem constants ----------------
#define BB 4
#define SS 1024
#define DD 1024
#define HH 16
#define DQ 64
#define DV 32
#define MASK_VAL -10000.0f

typedef __attribute__((ext_vector_type(16))) __bf16 v16bf;
typedef __attribute__((ext_vector_type(8)))  float  v8f;

__device__ __forceinline__ unsigned short f2bf(float f) {
  __bf16 h = (__bf16)f;
  return __builtin_bit_cast(unsigned short, h);
}
__device__ __forceinline__ float bf2f(unsigned short u) {
  __bf16 h = __builtin_bit_cast(__bf16, u);
  return (float)h;
}

__device__ __forceinline__ v8f wmma_bf16(v16bf a, v16bf b, v8f c) {
  // D = A(16x32 bf16) * B(32x16 bf16) + C(16x16 f32)
  return __builtin_amdgcn_wmma_f32_16x16x32_bf16(
      false, a, false, b, (short)0, c, false, false);
}

union AOp { v16bf v; unsigned int u[8]; };

// K-dim index of dword pair p (covers elements e=2p,2p+1) in the 16-bit
// A/B k order this code uses (contiguous): K(e) = e + (laneHi?16:0)
// Pairs p=0..7 cover K=[hi*16, hi*16+16),
// i.e. one contiguous 32-byte chunk per lane.
__device__ __forceinline__ int a_pair_k(int p, int laneHi) {
  return 2 * p + (laneHi ? 16 : 0);
}

// ---------------- stage 1: f32 -> bf16 converts ----------------
__global__ void convert_plain(const float* __restrict__ in,
                              unsigned short* __restrict__ out, int n) {
  int i = blockIdx.x * blockDim.x + threadIdx.x;
  if (i < n) out[i] = f2bf(in[i]);
}

// in: [R,C] row-major f32 -> out: [C,R] row-major bf16 (i.e. transpose)
__global__ void convert_T(const float* __restrict__ in,
                          unsigned short* __restrict__ out, int R, int C) {
  int i = blockIdx.x * blockDim.x + threadIdx.x;
  int n = R * C;
  if (i < n) {
    int r = i / C;
    int c = i - r * C;
    out[c * R + r] = f2bf(in[i]);
  }
}

// ---------------- stage 2/4: generic bf16 WMMA GEMM ----------------
// C[M,N] = A[M,K](bf16 row-major) * Bt[N,K]^T (bf16, pre-transposed) + bias[N]
// Per wave: 16(M) x 64(N) output tile -> 1 A-operand load feeds 4 WMMAs.
// mode 0: store bf16 row-major [M,N]
// mode 1: store bf16 "V-transposed": out[((m>>10)*N + n)*1024 + (m&1023)]
// mode 2: store f32  row-major [M,N]
__global__ __launch_bounds__(128)
void gemm_bf16_wmma(const unsigned short* __restrict__ A,
                    const unsigned short* __restrict__ Bt,
                    const float* __restrict__ bias,
                    void* __restrict__ out,
                    int M, int N, int K, int mode) {
  const int ntilesN = N >> 6;  // 64-wide N tiles
  const int ntiles  = (M >> 4) * ntilesN;
  const int wave = (blockIdx.x * (blockDim.x >> 5)) + (threadIdx.x >> 5);
  if (wave >= ntiles) return;  // wave-uniform

  const int mt = wave / ntilesN, nt = wave - mt * ntilesN;
  const int m0 = mt << 4, n0 = nt << 6;
  const int lane = threadIdx.x & 31;
  const int lm = lane & 15;
  const int hi = lane >> 4;

  const unsigned short* Arow = A + (size_t)(m0 + lm) * K;
  const unsigned short* Brow0 = Bt + (size_t)(n0 + lm) * K + hi * 16;
  const size_t bstride16 = (size_t)16 * K;  // 16 N-columns ahead

  v8f acc0 = {}, acc1 = {}, acc2 = {}, acc3 = {};
  const int nk = K >> 5;
  for (int kc = 0; kc < nk; ++kc) {
    AOp a;
#pragma unroll
    for (int p = 0; p < 8; ++p) {
      int kp = a_pair_k(p, hi);
      a.u[p] = *(const unsigned int*)(Arow + kc * 32 + kp);
    }
    const unsigned short* bp = Brow0 + kc * 32;
    __builtin_prefetch(bp + 32, 0, 0);  // global_prefetch_b8, next K-chunk
    v16bf b0 = *(const v16bf*)(bp);
    v16bf b1 = *(const v16bf*)(bp + bstride16);
    v16bf b2 = *(const v16bf*)(bp + 2 * bstride16);
    v16bf b3 = *(const v16bf*)(bp + 3 * bstride16);
    acc0 = wmma_bf16(a.v, b0, acc0);
    acc1 = wmma_bf16(a.v, b1, acc1);
    acc2 = wmma_bf16(a.v, b2, acc2);
    acc3 = wmma_bf16(a.v, b3, acc3);
  }

  v8f accs[4] = {acc0, acc1, acc2, acc3};
#pragma unroll
  for (int j = 0; j < 4; ++j) {
    const int ncol = n0 + j * 16 + lm;
    const float bval = bias[ncol];
#pragma unroll
    for (int r = 0; r < 8; ++r) {
      const int mrow = m0 + r + hi * 8;
      const float val = accs[j][r] + bval;
      if (mode == 0) {
        ((unsigned short*)out)[(size_t)mrow * N + ncol] = f2bf(val);
      } else if (mode == 1) {
        ((unsigned short*)out)[((size_t)(mrow >> 10) * N + ncol) * 1024 +
                               (mrow & 1023)] = f2bf(val);
      } else {
        ((float*)out)[(size_t)mrow * N + ncol] = val;
      }
    }
  }
}

// ---------------- stage 3: bidirectional attention ----------------
// One wave (32 threads) per (b, h, 16-row q-tile).
// Qb, Kb: bf16 [B,S,H*DQ]; Vt: bf16 [B, H*DV, S]
// prob_out: f32 at d_out+B*S*D, layout [B,H,S,2S] (fwd | bwd)
// catb: bf16 [B,S, H*DV*2] = [fwd_out | bwd_out]
__global__ __launch_bounds__(32)
void biattn_wmma(const unsigned short* __restrict__ Qb,
                 const unsigned short* __restrict__ Kb,
                 const unsigned short* __restrict__ Vt,
                 float* __restrict__ prob_out,
                 unsigned short* __restrict__ catb) {
  __shared__ unsigned short sc[16 * SS];  // score tile (bf16), 32 KB
  __shared__ float red0[32], red1[32];
  __shared__ float mfv[16], rsfv[16], mbv[16], rsbv[16];

  const int id = blockIdx.x;
  const int qt = id & 63;
  const int h  = (id >> 6) & 15;
  const int b  = id >> 10;

  const int lane = threadIdx.x & 31;
  const int lm = lane & 15;
  const int hi = lane >> 4;

  // ---- preload Q tile as two A-operands (d-chunks of 32) ----
  const unsigned short* Qrow =
      Qb + ((size_t)(b * SS + qt * 16 + lm) * (HH * DQ)) + h * DQ;
  AOp aq0, aq1;
#pragma unroll
  for (int p = 0; p < 8; ++p) {
    int kp = a_pair_k(p, hi);
    aq0.u[p] = *(const unsigned int*)(Qrow + kp);
    aq1.u[p] = *(const unsigned int*)(Qrow + 32 + kp);
  }

  // ---- scores: 64 key tiles, 2 WMMA each ----
  for (int kt = 0; kt < 64; ++kt) {
    const unsigned short* Krow =
        Kb + ((size_t)(b * SS + kt * 16 + lm) * (HH * DQ)) + h * DQ + hi * 16;
    __builtin_prefetch(Krow + HH * DQ * 16, 0, 0);  // next key tile
    v16bf b0 = *(const v16bf*)(Krow);
    v16bf b1 = *(const v16bf*)(Krow + 32);
    v8f c = {};
    c = wmma_bf16(aq0.v, b0, c);
    c = wmma_bf16(aq1.v, b1, c);
#pragma unroll
    for (int r = 0; r < 8; ++r) {
      const int M = r + hi * 8;
      sc[M * SS + kt * 16 + lm] = f2bf(c[r] * 0.125f);  // 1/sqrt(64)
    }
  }
  __syncthreads();

  // ---- softmax stats: 2 threads per row (halves of 512 cols) ----
  {
    const int row = lm;
    const int qrow = qt * 16 + row;
    const int c0 = hi * 512;
    float pmf = MASK_VAL, pmb = MASK_VAL;
    for (int c = 0; c < 512; ++c) {
      const int col = c0 + c;
      const float s = bf2f(sc[row * SS + col]);
      if (col <= qrow && s > pmf) pmf = s;
      if (col >= qrow && s > pmb) pmb = s;
    }
    red0[lane] = pmf; red1[lane] = pmb;
    __syncthreads();
    if (lane < 16) {
      mfv[lane] = fmaxf(red0[lane], red0[lane + 16]);
      mbv[lane] = fmaxf(red1[lane], red1[lane + 16]);
    }
    __syncthreads();
    const float mf = mfv[row], mb = mbv[row];
    float ssf = 0.f, ssb = 0.f;
    for (int c = 0; c < 512; ++c) {
      const int col = c0 + c;
      const float s = bf2f(sc[row * SS + col]);
      ssf += (col <= qrow) ? __expf(s - mf) : 0.f;
      ssb += (col >= qrow) ? __expf(s - mb) : 0.f;
    }
    red0[lane] = ssf; red1[lane] = ssb;
    __syncthreads();
    if (lane < 16) {
      rsfv[lane] = 1.0f / (red0[lane] + red0[lane + 16]);
      rsbv[lane] = 1.0f / (red1[lane] + red1[lane + 16]);
    }
    __syncthreads();
  }

  // ---- stream attn_prob to HBM (coalesced, dominant traffic) ----
  {
    for (int idx = lane; idx < 16 * SS; idx += 32) {
      const int row = idx >> 10;
      const int col = idx & 1023;
      const int qrow = qt * 16 + row;
      const float s = bf2f(sc[row * SS + col]);
      const float pf =
          (col <= qrow) ? __expf(s - mfv[row]) * rsfv[row] : 0.f;
      const float pb =
          (col >= qrow) ? __expf(s - mbv[row]) * rsbv[row] : 0.f;
      const size_t base =
          ((size_t)((b * HH + h) * SS + qrow)) * (2 * SS);
      prob_out[base + col] = pf;
      prob_out[base + SS + col] = pb;
    }
  }

  // ---- P * V : 32 key-chunks x (fwd,bwd) x (2 N-halves of DV) ----
  v8f cf0 = {}, cf1 = {}, cb0 = {}, cb1 = {};
  const unsigned short* V0 =
      Vt + ((size_t)(b * (HH * DV) + h * DV + lm) * SS) + hi * 16;
  const unsigned short* V1 = V0 + (size_t)16 * SS;
  for (int ch = 0; ch < 32; ++ch) {
    AOp af, ab;
    const int row = lm;
    const int qrow = qt * 16 + row;
    const float mf = mfv[row], rf = rsfv[row];
    const float mb = mbv[row], rb = rsbv[row];
#pragma unroll
    for (int e = 0; e < 16; ++e) {
      const int kk = e + (hi ? 16 : 0);
      const int col = ch * 32 + kk;
      const float s = bf2f(sc[row * SS + col]);
      const float pf = (col <= qrow) ? __expf(s - mf) * rf : 0.f;
      const float pb = (col >= qrow) ? __expf(s - mb) * rb : 0.f;
      af.v[e] = (__bf16)pf;
      ab.v[e] = (__bf16)pb;
    }
    v16bf b0 = *(const v16bf*)(V0 + ch * 32);
    v16bf b1 = *(const v16bf*)(V1 + ch * 32);
    cf0 = wmma_bf16(af.v, b0, cf0);
    cf1 = wmma_bf16(af.v, b1, cf1);
    cb0 = wmma_bf16(ab.v, b0, cb0);
    cb1 = wmma_bf16(ab.v, b1, cb1);
  }

  // ---- store concat tile: [B,S, fwd(H*DV) | bwd(H*DV)] ----
#pragma unroll
  for (int r = 0; r < 8; ++r) {
    const int srow = qt * 16 + r + hi * 8;
    const size_t rowbase = (size_t)(b * SS + srow) * (2 * HH * DV);
    catb[rowbase + h * DV + lm]            = f2bf(cf0[r]);
    catb[rowbase + h * DV + 16 + lm]       = f2bf(cf1[r]);
    catb[rowbase + HH * DV + h * DV + lm]      = f2bf(cb0[r]);
    catb[rowbase + HH * DV + h * DV + 16 + lm] = f2bf(cb1[r]);
  }
}

// ---------------- host-side orchestration ----------------
extern "C" void kernel_launch(void* const* d_in, const int* in_sizes, int n_in,
                              void* d_out, int out_size, void* d_ws,
                              size_t ws_size, hipStream_t stream) {
  (void)in_sizes; (void)n_in; (void)out_size; (void)ws_size;

  const float* x  = (const float*)d_in[0];  // [B,S,D]
  const float* Wq = (const float*)d_in[1];  // [D, H*DQ]
  const float* bq = (const float*)d_in[2];
  const float* Wk = (const float*)d_in[3];
  const float* bk = (const float*)d_in[4];
  const float* Wv = (const float*)d_in[5];  // [D, H*DV]
  const float* bv = (const float*)d_in[6];
  const float* Wo = (const float*)d_in[7];  // [D, D]
  const float* bo = (const float*)d_in[8];

  char* ws = (char*)d_ws;
  const size_t MB = 1024 * 1024;
  unsigned short* xb   = (unsigned short*)(ws + 0 * MB);   // 8 MB  [4096,1024]
  unsigned short* WqT  = (unsigned short*)(ws + 8 * MB);   // 2 MB  [1024,1024]
  unsigned short* WkT  = (unsigned short*)(ws + 10 * MB);  // 2 MB
  unsigned short* WvT  = (unsigned short*)(ws + 12 * MB);  // 1 MB  [512,1024]
  unsigned short* WoT  = (unsigned short*)(ws + 13 * MB);  // 2 MB
  unsigned short* Qb   = (unsigned short*)(ws + 15 * MB);  // 8 MB  [4096,1024]
  unsigned short* Kb   = (unsigned short*)(ws + 23 * MB);  // 8 MB
  unsigned short* Vt   = (unsigned short*)(ws + 31 * MB);  // 4 MB  [B,512,1024]
  unsigned short* catb = (unsigned short*)(ws + 35 * MB);  // 8 MB  [4096,1024]

  float* out_f = (float*)d_out;                       // [B,S,D]
  float* prob  = out_f + (size_t)BB * SS * DD;        // [B,H,S,2S]

  const int M = BB * SS;  // 4096

  // stage 1: converts
  {
    int n = M * DD;
    convert_plain<<<(n + 255) / 256, 256, 0, stream>>>(x, xb, n);
    n = DD * (HH * DQ);
    convert_T<<<(n + 255) / 256, 256, 0, stream>>>(Wq, WqT, DD, HH * DQ);
    convert_T<<<(n + 255) / 256, 256, 0, stream>>>(Wk, WkT, DD, HH * DQ);
    n = DD * (HH * DV);
    convert_T<<<(n + 255) / 256, 256, 0, stream>>>(Wv, WvT, DD, HH * DV);
    n = DD * DD;
    convert_T<<<(n + 255) / 256, 256, 0, stream>>>(Wo, WoT, DD, DD);
  }

  // stage 2: QKV projections (bf16 WMMA GEMM, 16x64 tile per wave)
  {
    int tiles = (M / 16) * ((HH * DQ) / 64);
    int grid = (tiles + 3) / 4;
    gemm_bf16_wmma<<<grid, 128, 0, stream>>>(xb, WqT, bq, Qb, M, HH * DQ, DD, 0);
    gemm_bf16_wmma<<<grid, 128, 0, stream>>>(xb, WkT, bk, Kb, M, HH * DQ, DD, 0);
    tiles = (M / 16) * ((HH * DV) / 64);
    grid = (tiles + 3) / 4;
    gemm_bf16_wmma<<<grid, 128, 0, stream>>>(xb, WvT, bv, Vt, M, HH * DV, DD, 1);
  }

  // stage 3: bidirectional attention (+ attn_prob streaming)
  biattn_wmma<<<BB * HH * (SS / 16), 32, 0, stream>>>(Qb, Kb, Vt, prob, catb);

  // stage 4: output projection -> f32 d_out
  {
    int tiles = (M / 16) * (DD / 64);
    int grid = (tiles + 3) / 4;
    gemm_bf16_wmma<<<grid, 128, 0, stream>>>(catb, WoT, bo, out_f, M, DD, DD, 2);
  }
}
